// SingleHead_83081847373987
// MI455X (gfx1250) — matrix operand order repaired and checked by reading, not compiled, not run on hardware
//
#include <hip/hip_runtime.h>
#include <hip/hip_bf16.h>
#include <math.h>

typedef __attribute__((ext_vector_type(16))) __bf16 v16bf;
typedef __attribute__((ext_vector_type(8)))  __bf16 bf16x8;
typedef __attribute__((ext_vector_type(8)))  float  v8f;
typedef __attribute__((ext_vector_type(4)))  unsigned int v4u;
typedef __attribute__((ext_vector_type(8)))  int v8i;
typedef __attribute__((ext_vector_type(4)))  int v4i;

#define Bn 4
#define Tn 4096
#define Cn 512
#define Hn 64

#if defined(__AMDGCN__) && __has_builtin(__builtin_amdgcn_tensor_load_to_lds)
#define USE_TDM 1
#else
#define USE_TDM 0
#endif

static __device__ __forceinline__ v8f wmma_bf16(v16bf a, v16bf b, v8f c) {
  // (neg_a, A, neg_b, B, c_mod, C, reuse_a, reuse_b)
  return __builtin_amdgcn_wmma_f32_16x16x32_bf16(false, a, false, b, (short)0, c,
                                                 false, false);
}

// A-matrix 16x32 bf16 fragment (ISA 7.12.2): lanes 0-15 hold row=lane16 with
// K={base..base+7, base+16..base+23}; lanes 16-31 same rows, base shifted by 8
// (caller folds the +8 into `base`). `row` points at contiguous K-major data.
static __device__ __forceinline__ v16bf load_a_frag(const __bf16* row, int base) {
  bf16x8 lo = *(const bf16x8*)(row + base);
  bf16x8 hi = *(const bf16x8*)(row + base + 16);
  v16bf f;
#pragma unroll
  for (int i = 0; i < 8; ++i) { f[i] = lo[i]; f[i + 8] = hi[i]; }
  return f;
}

// B-matrix 32x16 bf16 fragment: lane L holds column N=L&15; lanes 0-15 carry
// K={k0..k0+7, k0+16..k0+23}, lanes 16-31 same with k0 shifted by 8 (caller folds +8 into k0).
static __device__ __forceinline__ v16bf load_b_frag(const __bf16* col, int k0) {
  bf16x8 lo = *(const bf16x8*)(col + k0);
  bf16x8 hi = *(const bf16x8*)(col + k0 + 16);
  v16bf f;
#pragma unroll
  for (int i = 0; i < 8; ++i) { f[i] = lo[i]; f[i + 8] = hi[i]; }
  return f;
}

#if USE_TDM
// 2D TDM load: tile_d1 rows of tile_d0 contiguous elements (bf16, data_size=1)
// packed into LDS at lds_off. D# layout per ISA 8.3-8.5.
// clang-23 (amdgpu-toolchain) builtin: 6 args (g0, g1, g2, g3, g4, cpol).
static __device__ __forceinline__ void tdm_load_2d(
    unsigned lds_off, const void* gaddr, unsigned tile_d0, unsigned tile_d1,
    unsigned tensor_d0, unsigned tensor_d1, unsigned stride0) {
  unsigned long long ga = (unsigned long long)(uintptr_t)gaddr;
  v4u g0;
  g0[0] = 1u;                                       // count=1, user descriptor
  g0[1] = lds_off;                                  // lds_addr (bytes)
  g0[2] = (unsigned)(ga & 0xffffffffu);             // global_addr[31:0]
  g0[3] = (unsigned)((ga >> 32) & 0x01ffffffu) | (2u << 30);  // addr[56:32]|type=2
  v8i g1;
  g1[0] = (int)(1u << 16);                          // wg_mask=0, data_size=1 (2B)
  g1[1] = (int)(tensor_d0 << 16);                   // tensor_dim0[15:0]
  g1[2] = (int)((tensor_d0 >> 16) | (tensor_d1 << 16));
  g1[3] = (int)((tensor_d1 >> 16) | (tile_d0 << 16));
  g1[4] = (int)(tile_d1 & 0xffffu);                 // tile_dim1; tile_dim2=0
  g1[5] = (int)stride0;                             // tensor_dim0_stride[31:0]
  g1[6] = 0;
  g1[7] = 0;                                        // tensor_dim1_stride unused
  v4i g2 = {0, 0, 0, 0};
  v4i g3 = {0, 0, 0, 0};
  v8i g4 = {0, 0, 0, 0, 0, 0, 0, 0};
  __builtin_amdgcn_tensor_load_to_lds(g0, g1, g2, g3, g4, 0);
}
#endif

// ------------- Weight prep: W[C,H] f32 -> WT[3][H][C] bf16 (L2 resident) -----
__global__ __launch_bounds__(256) void sh_wprep_kernel(
    const float* __restrict__ Wq, const float* __restrict__ Wk,
    const float* __restrict__ Wv, __bf16* __restrict__ WT) {
  const int per = Hn * Cn;
  const int idx = blockIdx.x * 256 + threadIdx.x;
  if (idx >= 3 * per) return;
  const int which = idx / per;
  const int rem = idx % per;
  const int h = rem / Cn;
  const int c = rem % Cn;
  const float* W = (which == 0) ? Wq : (which == 1) ? Wk : Wv;
  WT[idx] = (__bf16)W[(size_t)c * Hn + h];
}

// ---------------- Q/K/V projection: one wave -> 16x64 output tile ------------
__global__ __launch_bounds__(32) void sh_proj_kernel(
    const float* __restrict__ x, const __bf16* __restrict__ WT,
    __bf16* __restrict__ qb, __bf16* __restrict__ kb, __bf16* __restrict__ vT) {
  const int ln     = threadIdx.x;
  const int lane16 = ln & 15;
  const int half   = ln >> 4;
  const int abase  = half * 8;    // A-fragment K offset for this lane half
  const int bbase  = half * 8;    // B-fragment K offset for this lane half
  const int tbase  = blockIdx.x * 16;  // flat row over B*T
  const int which  = blockIdx.y;       // 0=Q 1=K 2=V
  const __bf16* wtb = WT + (size_t)which * Hn * Cn;

  v8f acc[4] = {};
  const float* xr = x + (size_t)(tbase + lane16) * Cn;

#pragma unroll 1
  for (int cb = 0; cb < Cn; cb += 32) {
    v16bf a;
#pragma unroll
    for (int i = 0; i < 8; ++i) {
      a[i]     = (__bf16)xr[cb + abase + i];
      a[i + 8] = (__bf16)xr[cb + abase + 16 + i];
    }
#pragma unroll
    for (int n = 0; n < 4; ++n) {
      const __bf16* wcol = wtb + (size_t)(n * 16 + lane16) * Cn + cb;
      acc[n] = wmma_bf16(a, load_b_frag(wcol, bbase), acc[n]);
    }
  }

  const float qscale = 0.04419417382415922f;  // 1/sqrt(C) folded into Q
#pragma unroll
  for (int n = 0; n < 4; ++n) {
    const int h = n * 16 + lane16;
#pragma unroll
    for (int r = 0; r < 8; ++r) {
      const int tr = tbase + r + 8 * half;     // C/D layout: row = r + 8*half
      const float val = acc[n][r];
      if (which == 0) {
        qb[(size_t)tr * Hn + h] = (__bf16)(val * qscale);
      } else if (which == 1) {
        kb[(size_t)tr * Hn + h] = (__bf16)val;
      } else {
        const int bi = tr >> 12;               // /Tn
        const int t  = tr & (Tn - 1);
        vT[((size_t)bi * Hn + h) * Tn + t] = (__bf16)val;  // V stored [B,H,T]
      }
    }
  }
}

// -------- Flash attention: one wave -> 16 query rows, stream 32-key blocks ---
__global__ __launch_bounds__(32) void sh_attn_kernel(
    const __bf16* __restrict__ qb, const __bf16* __restrict__ kbm,
    const __bf16* __restrict__ vT, float* __restrict__ out) {
  __shared__ __attribute__((aligned(16))) __bf16 k_lds[32 * Hn];  // [key][h]
  __shared__ __attribute__((aligned(16))) __bf16 v_lds[Hn * 32];  // [h][key]
  __shared__ __attribute__((aligned(16))) __bf16 p_lds[16 * 32];  // [q][key]

  const int ln     = threadIdx.x;
  const int lane16 = ln & 15;
  const int half   = ln >> 4;
  const int abase  = half * 8;
  const int bbase  = half * 8;
  const int bi     = blockIdx.x >> 8;          // Tn/16 = 256 tiles per batch
  const int q0     = (blockIdx.x & 255) * 16;

  const __bf16* qB = qb  + (size_t)bi * Tn * Hn;
  const __bf16* kB = kbm + (size_t)bi * Tn * Hn;
  const __bf16* vB = vT  + (size_t)bi * Hn * Tn;

  const __bf16* qrow = qB + (size_t)(q0 + lane16) * Hn;
  const v16bf aq0 = load_a_frag(qrow, abase);        // H chunk 0..31
  const v16bf aq1 = load_a_frag(qrow, 32 + abase);   // H chunk 32..63

#if USE_TDM
  const unsigned klds_off = (unsigned)(uintptr_t)&k_lds[0];
  const unsigned vlds_off = (unsigned)(uintptr_t)&v_lds[0];
#endif

  float m_run[8], l_run[8];
#pragma unroll
  for (int r = 0; r < 8; ++r) { m_run[r] = -__builtin_inff(); l_run[r] = 0.0f; }
  v8f oacc[4] = {};

  for (int j0 = 0; j0 < q0 + 16; j0 += 32) {
#if USE_TDM
    // WAR: previous iteration's DS reads must finish before TDM rewrites LDS
    asm volatile("s_wait_dscnt 0x0" ::: "memory");
    tdm_load_2d(klds_off, kB + (size_t)j0 * Hn, Hn, 32, Hn, Tn, Hn);
    tdm_load_2d(vlds_off, vB + j0, 32, Hn, Tn, Hn, Tn);
    __builtin_amdgcn_s_wait_tensorcnt(0);
    asm volatile("" ::: "memory");
    const __bf16* krA = k_lds + (size_t)lane16 * Hn;
    const __bf16* krB = k_lds + (size_t)(16 + lane16) * Hn;
#else
    if (j0 + 64 <= Tn)
      __builtin_prefetch(kB + (size_t)(j0 + 32 + lane16) * Hn, 0, 3);
    const __bf16* krA = kB + (size_t)(j0 + lane16) * Hn;
    const __bf16* krB = kB + (size_t)(j0 + 16 + lane16) * Hn;
#endif

    // S = Q * K^T for keys [j0, j0+32): two 16x16 f32 tiles
    v8f s0 = {}, s1 = {};
    s0 = wmma_bf16(aq0, load_b_frag(krA, bbase), s0);
    s0 = wmma_bf16(aq1, load_b_frag(krA, 32 + bbase), s0);
    s1 = wmma_bf16(aq0, load_b_frag(krB, bbase), s1);
    s1 = wmma_bf16(aq1, load_b_frag(krB, 32 + bbase), s1);

    // Causal mask in C/D layout: lane holds (row = r + 8*half, col = lane16)
    const int kc0 = j0 + lane16;
    const int kc1 = kc0 + 16;
    const int qr0 = q0 + 8 * half;
    if (j0 + 31 > q0) {
#pragma unroll
      for (int r = 0; r < 8; ++r) {
        const int qg = qr0 + r;
        if (kc0 > qg) s0[r] = -__builtin_inff();
        if (kc1 > qg) s1[r] = -__builtin_inff();
      }
    }

    // Online softmax: rowwise reductions across each 16-lane half
    float sc[8];
#pragma unroll
    for (int r = 0; r < 8; ++r) {
      float mt = fmaxf(s0[r], s1[r]);
      mt = fmaxf(mt, __shfl_xor(mt, 1, 32));
      mt = fmaxf(mt, __shfl_xor(mt, 2, 32));
      mt = fmaxf(mt, __shfl_xor(mt, 4, 32));
      mt = fmaxf(mt, __shfl_xor(mt, 8, 32));
      const float mn  = fmaxf(m_run[r], mt);
      const float scr = __expf(m_run[r] - mn);
      const float p0  = __expf(s0[r] - mn);
      const float p1  = __expf(s1[r] - mn);
      float rs = p0 + p1;
      rs += __shfl_xor(rs, 1, 32);
      rs += __shfl_xor(rs, 2, 32);
      rs += __shfl_xor(rs, 4, 32);
      rs += __shfl_xor(rs, 8, 32);
      l_run[r] = l_run[r] * scr + rs;
      m_run[r] = mn;
      sc[r]    = scr;
      // C/D layout -> row-major LDS (bf16) for the A-layout reload (transpose)
      p_lds[(8 * half + r) * 32 + lane16]      = (__bf16)p0;
      p_lds[(8 * half + r) * 32 + 16 + lane16] = (__bf16)p1;
    }

#pragma unroll
    for (int n = 0; n < 4; ++n)
#pragma unroll
      for (int r = 0; r < 8; ++r) oacc[n][r] = oacc[n][r] * sc[r];

    // Single wave per WG: same-wave DS ordering + explicit dscnt wait suffices
    asm volatile("s_wait_dscnt 0x0" ::: "memory");
    const v16bf pa = load_a_frag(p_lds + lane16 * 32, abase);

    // O += P * V ; V^T layout makes key-dim contiguous per H column
#pragma unroll
    for (int n = 0; n < 4; ++n) {
#if USE_TDM
      const __bf16* vcol = v_lds + (size_t)(n * 16 + lane16) * 32;
#else
      const __bf16* vcol = vB + (size_t)(n * 16 + lane16) * Tn + j0;
#endif
      oacc[n] = wmma_bf16(pa, load_b_frag(vcol, bbase), oacc[n]);
    }
  }

  float* outB = out + (size_t)bi * Tn * Hn;
#pragma unroll
  for (int r = 0; r < 8; ++r) {
    const float il = 1.0f / l_run[r];
    const int tr = q0 + 8 * half + r;
#pragma unroll
    for (int n = 0; n < 4; ++n)
      outB[(size_t)tr * Hn + n * 16 + lane16] = oacc[n][r] * il;
  }
}

extern "C" void kernel_launch(void* const* d_in, const int* in_sizes, int n_in,
                              void* d_out, int out_size, void* d_ws, size_t ws_size,
                              hipStream_t stream) {
  (void)in_sizes; (void)n_in; (void)out_size; (void)ws_size;
  const float* x  = (const float*)d_in[0];
  const float* Wq = (const float*)d_in[1];
  const float* Wk = (const float*)d_in[2];
  const float* Wv = (const float*)d_in[3];
  float* out = (float*)d_out;

  const size_t N = (size_t)Bn * Tn * Hn;        // 1,048,576 elems per tensor
  __bf16* qb = (__bf16*)d_ws;                   // [B,T,H] bf16 (pre-scaled)
  __bf16* kb = qb + N;                          // [B,T,H] bf16
  __bf16* vT = kb + N;                          // [B,H,T] bf16 (transposed)
  __bf16* WT = vT + N;                          // [3,H,C] bf16 weightsT

  const int wtotal = 3 * Hn * Cn;
  sh_wprep_kernel<<<dim3((wtotal + 255) / 256), dim3(256), 0, stream>>>(Wq, Wk, Wv, WT);
  dim3 pg((Bn * Tn) / 16, 3);
  sh_proj_kernel<<<pg, dim3(32), 0, stream>>>(x, WT, qb, kb, vT);
  sh_attn_kernel<<<dim3(Bn * (Tn / 16)), dim3(32), 0, stream>>>(qb, kb, vT, out);
}
